// LinearDynamics_39109972198112
// MI455X (gfx1250) — matrix operand order repaired and checked by reading, not compiled, not run on hardware
//
#include <hip/hip_runtime.h>

// ---------------------------------------------------------------------------
// z_{t+1} = z_t @ A ; collect all 256 states -> out[B=256, T=256, D=512] f32
//
// MI455X (gfx1250, wave32) design:
//  * 16 workgroups; each owns 16 rows of B and carries its 16x512 state in
//    LDS (bf16, ping-pong, 2x16KB) across all 256 sequential steps (the step
//    chain caps usable parallelism, so per-WGP efficiency is what matters).
//  * 8 waves/WG x 4 N-tiles/wave: the full-z LDS broadcast (16KB per wave
//    per step) is amortized over 4 tiles -> ~128KB/step LDS, balanced with
//    the ~512 WMMA/step matrix-pipe work per WGP.
//  * A is pre-packed once into bf16 WMMA B-operand fragments in d_ws; it is
//    L2-resident (512KB << 192MB) and re-streamed each step. The K-block
//    order is rotated by t so the loads are NOT loop-invariant: this stops
//    the compiler from hoisting 512 VGPRs of fragments and spilling to
//    scratch inside the hot loop (seen in the previous round's asm).
//  * v_wmma_f32_16x16x32_bf16 with f32 accumulation; output stream to HBM
//    (134MB total, ~5.8us floor at 23.3 TB/s) is the only real HBM traffic.
// ---------------------------------------------------------------------------

#define D_DIM   512
#define B_DIM   256
#define T_STEPS 256
#define KB      16               // 512 / 32 K-blocks
#define NTILES  32               // 512 / 16 N-tiles
#define WAVES   8
#define THREADS (WAVES * 32)     // 256 threads, wave32
#define NT      (NTILES / WAVES) // 4 N-tiles per wave

typedef __attribute__((ext_vector_type(16))) __bf16 v16bf;
typedef __attribute__((ext_vector_type(8)))  float  v8f;

__device__ __forceinline__ __bf16 f2bf(float f) {
  // round-to-nearest-even f32 -> bf16
  unsigned u = __builtin_bit_cast(unsigned, f);
  unsigned r = u + 0x7FFFu + ((u >> 16) & 1u);
  unsigned short h = (unsigned short)(r >> 16);
  return __builtin_bit_cast(__bf16, h);
}

// ---------------------------------------------------------------------------
// Pre-pack A (f32 row-major 512x512) into bf16 WMMA B-operand fragments.
// Fragment f = kb*32 + nb covers A[kb*32 .. kb*32+31][nb*16 .. nb*16+15].
// B-operand 16-bit layout (32x16): lanes 0-15 -> N=lane, lanes 16-31 -> N=lane-16;
// half j holds K = kb*32 + 8*(lane>=16) + (j&7) + (j>=8 ? 16 : 0).
// Stored lane-contiguous: P[f*512 + lane*16 + j].
// ---------------------------------------------------------------------------
__global__ void prepack_A_kernel(const float* __restrict__ A,
                                 __bf16* __restrict__ P) {
  int idx  = blockIdx.x * blockDim.x + threadIdx.x;   // 0 .. 512*512-1
  int f    = idx >> 9;                                // fragment id
  int e    = idx & 511;
  int lane = e >> 4;
  int j    = e & 15;
  int kb   = f >> 5;
  int nb   = f & 31;
  int n    = nb * 16 + (lane & 15);
  int k    = kb * 32 + ((lane < 16) ? 0 : 8) + (j & 7) + ((j >= 8) ? 16 : 0);
  P[idx] = f2bf(A[k * D_DIM + n]);
}

// ---------------------------------------------------------------------------
// Epilogue for one 16x16 output tile: stream f32 to global out, and scatter
// bf16 into the next-step LDS state in WMMA A-operand fragment layout.
// A-operand 16-bit layout (16x32): lane<16 holds K in {0..7,16..23},
// lane>=16 holds K in {8..15,24..31}; half j = (K&7) + (K>=16 ? 8 : 0).
// C/D layout: VGPR v, lanes 0-15 -> M=v, lanes 16-31 -> M=v+8, N=lane&15.
// ---------------------------------------------------------------------------
__device__ __forceinline__ void tile_epilogue(const v8f acc, int n, int lane,
                                              int rb, int t,
                                              float* __restrict__ out,
                                              __bf16* __restrict__ znext) {
  const int kl   = n & 31;
  const int kb   = n >> 5;
  const int base = kb * 512 + ((kl >> 3) & 1) * 256
                 + (kl & 7) + ((kl >= 16) ? 8 : 0);
  const int mo   = (lane < 16) ? 0 : 8;
  float* o = out + (size_t)(rb + mo) * (T_STEPS * D_DIM)
                 + (size_t)t * D_DIM + n;
#pragma unroll
  for (int v = 0; v < 8; ++v) {
    o[(size_t)v * (T_STEPS * D_DIM)] = acc[v];
    znext[base + (v + mo) * 16]      = f2bf(acc[v]);
  }
}

__global__ __launch_bounds__(THREADS)
void lindyn_kernel(const float* __restrict__ z0,
                   const __bf16* __restrict__ P,
                   float* __restrict__ out) {
  // ping-pong state: 16 A-fragments (16x32 bf16 = 512 halves) each = 16KB/buf
  __shared__ __bf16 zfrag[2][KB * 512];

  const int tid  = threadIdx.x;
  const int lane = tid & 31;
  const int wave = tid >> 5;
  const int rb   = blockIdx.x * 16;       // rows of B owned by this WG

  // ---- initial state: z0 rows -> A-fragment layout in buffer 0 ----
  for (int e = tid; e < 16 * D_DIM; e += THREADS) {
    int   m  = e >> 9;
    int   n  = e & 511;
    float vv = z0[(size_t)(rb + m) * D_DIM + n];
    int   kb = n >> 5;
    int   kl = n & 31;
    int   hi = (kl >> 3) & 1;
    int   j  = (kl & 7) + ((kl >= 16) ? 8 : 0);
    zfrag[0][kb * 512 + (m + hi * 16) * 16 + j] = f2bf(vv);
  }
  __syncthreads();

  const int nb0 = wave * NT;              // this wave's four N-tiles
  const int n0  = nb0 * 16 + (lane & 15);

  for (int t = 0; t < T_STEPS; ++t) {
    const int cur = t & 1;
    const int nxt = cur ^ 1;

    v8f acc0 = {}, acc1 = {}, acc2 = {}, acc3 = {};

    // K-block order rotated by t: same reduction set, but addresses are not
    // loop-invariant, so B-fragments stream from L2 instead of being hoisted
    // into (and spilled out of) registers.
#pragma unroll
    for (int kk = 0; kk < KB; ++kk) {
      const int kbe = (kk + t) & (KB - 1);
      v16bf af = *(const v16bf*)&zfrag[cur][kbe * 512 + lane * 16];
      const __bf16* pb = P + ((kbe * 32 + nb0) * 512) + lane * 16;
      v16bf b0 = *(const v16bf*)(pb);
      v16bf b1 = *(const v16bf*)(pb + 512);
      v16bf b2 = *(const v16bf*)(pb + 1024);
      v16bf b3 = *(const v16bf*)(pb + 1536);
      acc0 = __builtin_amdgcn_wmma_f32_16x16x32_bf16(
                 false, af, false, b0, (short)0, acc0, false, false);
      acc1 = __builtin_amdgcn_wmma_f32_16x16x32_bf16(
                 false, af, false, b1, (short)0, acc1, false, false);
      acc2 = __builtin_amdgcn_wmma_f32_16x16x32_bf16(
                 false, af, false, b2, (short)0, acc2, false, false);
      acc3 = __builtin_amdgcn_wmma_f32_16x16x32_bf16(
                 false, af, false, b3, (short)0, acc3, false, false);
    }

    tile_epilogue(acc0, n0,      lane, rb, t, out, &zfrag[nxt][0]);
    tile_epilogue(acc1, n0 + 16, lane, rb, t, out, &zfrag[nxt][0]);
    tile_epilogue(acc2, n0 + 32, lane, rb, t, out, &zfrag[nxt][0]);
    tile_epilogue(acc3, n0 + 48, lane, rb, t, out, &zfrag[nxt][0]);

    __syncthreads();   // next step reads zfrag[nxt]
  }
}

// ---------------------------------------------------------------------------
// Harness entry. Inputs: d_in[0]=z0 (256x512 f32), d_in[1]=A (512x512 f32),
// d_in[2]=steps (int, fixed 256 per reference). d_out: 256*256*512 f32.
// d_ws: 512KB used for pre-packed bf16 A fragments.
// ---------------------------------------------------------------------------
extern "C" void kernel_launch(void* const* d_in, const int* in_sizes, int n_in,
                              void* d_out, int out_size, void* d_ws,
                              size_t ws_size, hipStream_t stream) {
  (void)in_sizes; (void)n_in; (void)out_size; (void)ws_size;
  const float* z0 = (const float*)d_in[0];
  const float* A  = (const float*)d_in[1];
  __bf16*      P  = (__bf16*)d_ws;
  float*       out = (float*)d_out;

  prepack_A_kernel<<<(D_DIM * D_DIM) / 256, 256, 0, stream>>>(A, P);
  lindyn_kernel<<<B_DIM / 16, THREADS, 0, stream>>>(z0, P, out);
}
